// CharRNN_86792699118063
// MI455X (gfx1250) — matrix operand order repaired and checked by reading, not compiled, not run on hardware
//
#include <hip/hip_runtime.h>

#define B_N  128
#define SEQ  1024
#define HID  512
#define VOC  96
#define EMB  64
#define HSTR 520   // padded bf16 row stride for h buffers (bank-conflict avoidance)
#define XSTR 516   // padded f32 row stride for xh tile

typedef __attribute__((ext_vector_type(16))) __bf16 bf16x16;
typedef __attribute__((ext_vector_type(8)))  float  f32x8;

union Frag {
  uint4   u[2];
  bf16x16 v;
};

__device__ __forceinline__ unsigned short f2bf(float f) {
  unsigned u = __float_as_uint(f);
  u += 0x7fffu + ((u >> 16) & 1u);       // round-to-nearest-even
  return (unsigned short)(u >> 16);
}
__device__ __forceinline__ float bf2f(unsigned short h) {
  return __uint_as_float(((unsigned)h) << 16);
}

// Branch-free tanh from native v_exp_f32 (exp2) + v_rcp_f32:
//   tanh(x) = sign(x) * (1 - e) / (1 + e),  e = exp(-2|x|) = exp2(-2*log2(e)*|x|)
// Saturates correctly for large |x| (e -> 0), no EXEC divergence.
__device__ __forceinline__ float fast_tanh(float x) {
  float ax = __builtin_fabsf(x);
  float e  = __builtin_amdgcn_exp2f(ax * -2.8853900817779268f);
  float r  = (1.0f - e) * __builtin_amdgcn_rcpf(1.0f + e);
  return __builtin_copysignf(r, x);
}

// -------- prep 0: combined[v][h] = embedding[v] @ W_xh + b_h  (96 x 512 f32) --------
__global__ __launch_bounds__(256) void prep_combined(
    const float* __restrict__ emb, const float* __restrict__ wxh,
    const float* __restrict__ bh, float* __restrict__ combined) {
  int i = blockIdx.x * 256 + threadIdx.x;
  if (i >= VOC * HID) return;
  int v = i >> 9, h = i & (HID - 1);
  float acc = bh[h];
  #pragma unroll 8
  for (int e = 0; e < EMB; ++e) acc += emb[v * EMB + e] * wxh[e * HID + h];
  combined[i] = acc;
}

// -------- prep 1: pack f32 [512][N] row-major into bf16 WMMA B-fragment tiles --------
// tile(ks, nt) = 32x16 (K x N), stored as [lane][8 dwords]:
//   lanes 0-15 (col = lane):      dword v<4 holds K = 2v, 2v+1; v>=4 holds K = 8+2v, 9+2v
//   lanes 16-31 (col = lane-16):  dword v<4 holds K = 8+2v, 9+2v; v>=4 holds K = 16+2v, 17+2v
__global__ __launch_bounds__(256) void prep_pack(
    const float* __restrict__ W, unsigned* __restrict__ dst, int N, int ntiles) {
  int gid = blockIdx.x * 256 + threadIdx.x;
  if (gid >= 16 * ntiles * 32) return;
  int lane = gid & 31, tile = gid >> 5;
  int nt = tile % ntiles, ks = tile / ntiles;
  int lm = lane & 15;
  int khalf = (lane >= 16) ? 8 : 0;
  int n = nt * 16 + lm;
  unsigned* p = dst + (size_t)tile * 256 + lane * 8;
  #pragma unroll
  for (int v = 0; v < 8; ++v) {
    int k = ks * 32 + khalf + 2 * v + (v >= 4 ? 8 : 0);
    unsigned short lo = f2bf(W[(size_t)k * N + n]);
    unsigned short hh = f2bf(W[(size_t)(k + 1) * N + n]);
    p[v] = (unsigned)lo | ((unsigned)hh << 16);
  }
}

// -------- main fused persistent kernel: recurrence + fc, 8 WGs x 16 batch rows --------
__global__ __launch_bounds__(256) void rnn_fused(
    const int* __restrict__ x, const float* __restrict__ h0,
    const float* __restrict__ combined,
    const unsigned* __restrict__ whh_t, const unsigned* __restrict__ fcw_t,
    const float* __restrict__ fc_b,
    float* __restrict__ logits, float* __restrict__ finalh) {
  __shared__ __align__(16) unsigned short hbuf[2][16 * HSTR];  // bf16 h, double-buffered
  __shared__ __align__(16) float xh[16 * XSTR];                // staged input projection
  __shared__ int   idxs[16];
  __shared__ float fcbs[VOC];

  const int tid  = threadIdx.x;
  const int lane = tid & 31;
  const int wave = tid >> 5;
  const int lm   = lane & 15;
  const int hi8  = (lane >= 16) ? 8 : 0;   // A-frag K offset AND C-layout row offset
  const int rowbase = blockIdx.x * 16;
  const int ncb = wave * 64;               // this wave's 4 column tiles

  if (tid < VOC) fcbs[tid] = fc_b[tid];
  for (int i = tid; i < 16 * HID; i += 256) {
    int m = i >> 9, n = i & (HID - 1);
    hbuf[0][m * HSTR + n] = f2bf(h0[(rowbase + m) * HID + n]);
  }
  __syncthreads();

  for (int t = 0; t < SEQ; ++t) {
    const unsigned short* __restrict__ hcur = hbuf[t & 1];
    unsigned short* __restrict__ hnx = hbuf[(t + 1) & 1];

    if (tid < 16) idxs[tid] = x[(rowbase + tid) * SEQ + t];
    __syncthreads();
    for (int i = tid; i < 16 * HID; i += 256) {       // xh_t = combined[x_t] (+b_h folded)
      int m = i >> 9, n = i & (HID - 1);
      xh[m * XSTR + n] = combined[idxs[m] * HID + n];
    }
    __syncthreads();

    // ---- recurrence GEMM: acc = xh + h_{t-1} @ W_hh ----
    f32x8 acc[4];
    #pragma unroll
    for (int nt = 0; nt < 4; ++nt)
      #pragma unroll
      for (int r = 0; r < 8; ++r)
        acc[nt][r] = xh[(hi8 + r) * XSTR + ncb + nt * 16 + lm];

    for (int ks = 0; ks < 16; ++ks) {
      Frag a;
      const unsigned short* ap = hcur + lm * HSTR + ks * 32 + hi8;
      a.u[0] = *(const uint4*)ap;           // K = hi8 .. hi8+7
      a.u[1] = *(const uint4*)(ap + 16);    // K = 16+hi8 .. 16+hi8+7
      #pragma unroll
      for (int nt = 0; nt < 4; ++nt) {
        Frag b;
        const uint4* bp =
            (const uint4*)(whh_t + ((size_t)(ks * 32 + (ncb >> 4) + nt)) * 256 + lane * 8);
        b.u[0] = bp[0];
        b.u[1] = bp[1];
        acc[nt] = __builtin_amdgcn_wmma_f32_16x16x32_bf16(
            false, a.v, false, b.v, (short)0, acc[nt], false, false);
      }
    }

    #pragma unroll
    for (int nt = 0; nt < 4; ++nt)
      #pragma unroll
      for (int r = 0; r < 8; ++r)
        hnx[(hi8 + r) * HSTR + ncb + nt * 16 + lm] = f2bf(fast_tanh(acc[nt][r]));
    __syncthreads();

    // ---- fused output FC: logits_t = h_t @ fc_w + fc_b (6 tiles, waves 0..5) ----
    if (wave < 6) {
      f32x8 cf;
      #pragma unroll
      for (int r = 0; r < 8; ++r) cf[r] = fcbs[wave * 16 + lm];
      for (int ks = 0; ks < 16; ++ks) {
        Frag a, b;
        const unsigned short* ap = hnx + lm * HSTR + ks * 32 + hi8;
        a.u[0] = *(const uint4*)ap;
        a.u[1] = *(const uint4*)(ap + 16);
        const uint4* bp = (const uint4*)(fcw_t + ((size_t)(ks * 6 + wave)) * 256 + lane * 8);
        b.u[0] = bp[0];
        b.u[1] = bp[1];
        cf = __builtin_amdgcn_wmma_f32_16x16x32_bf16(
            false, a.v, false, b.v, (short)0, cf, false, false);
      }
      #pragma unroll
      for (int r = 0; r < 8; ++r) {
        size_t o = ((size_t)(rowbase + hi8 + r) * SEQ + t) * VOC + wave * 16 + lm;
        logits[o] = cf[r];
      }
    }
  }

  __syncthreads();
  for (int i = tid; i < 16 * HID; i += 256) {   // h after t=1023 lives in hbuf[0]
    int m = i >> 9, n = i & (HID - 1);
    finalh[(rowbase + m) * HID + n] = bf2f(hbuf[0][m * HSTR + n]);
  }
}

extern "C" void kernel_launch(void* const* d_in, const int* in_sizes, int n_in,
                              void* d_out, int out_size, void* d_ws, size_t ws_size,
                              hipStream_t stream) {
  const int*   x   = (const int*)  d_in[0];
  const float* hid = (const float*)d_in[1];
  const float* emb = (const float*)d_in[2];
  const float* wxh = (const float*)d_in[3];
  const float* whh = (const float*)d_in[4];
  const float* bh  = (const float*)d_in[5];
  const float* fcw = (const float*)d_in[6];
  const float* fcb = (const float*)d_in[7];

  char* ws = (char*)d_ws;
  float*    combined = (float*)ws;                           // 96*512*4   = 196608 B
  unsigned* whh_t    = (unsigned*)(ws + 196608);             // 512*512*2  = 524288 B
  unsigned* fcw_t    = (unsigned*)(ws + 196608 + 524288);    // 512*96*2   =  98304 B

  prep_combined<<<(VOC * HID + 255) / 256, 256, 0, stream>>>(emb, wxh, bh, combined);
  prep_pack<<<(16 * 32 * 32 + 255) / 256, 256, 0, stream>>>(whh, whh_t, HID, 32);
  prep_pack<<<(16 * 6 * 32 + 255) / 256, 256, 0, stream>>>(fcw, fcw_t, VOC, 6);

  float* logits = (float*)d_out;
  float* finalh = logits + (size_t)B_N * SEQ * VOC;
  rnn_fused<<<B_N / 16, 256, 0, stream>>>(x, hid, combined, whh_t, fcw_t, fcb,
                                          logits, finalh);
}
